// OuterProductMean_58918361366846
// MI455X (gfx1250) — matrix operand order repaired and checked by reading, not compiled, not run on hardware
//
#include <hip/hip_runtime.h>

typedef _Float16 half_t;
typedef __attribute__((ext_vector_type(16))) _Float16 v16h;
typedef __attribute__((ext_vector_type(8)))  _Float16 v8h;
typedef __attribute__((ext_vector_type(8)))  float    v8f;

#define IN_DIM 128
#define DIM    32
#define SEQ    256   // i/j extent
#define RROWS  128   // reduction rows
#define TI     4     // i per block  -> GEMM1 tile 128 rows
#define TJ     4     // j per block  -> GEMM1 tile 128 cols

// LDS row strides (halfs), padded so 16 lanes' ds_load_b128 hit distinct banks
#define ASTRIDE 136   // 128 + 8  (272B rows, 16B aligned)
#define OSTRIDE 1032  // 1024 + 8 (2064B rows, 16B aligned)

// gfx1250 async global->LDS path (VGPR-bypassing DMA, tracked by ASYNCcnt)
#if defined(__has_builtin)
#if __has_builtin(__builtin_amdgcn_global_load_async_to_lds_b128) && \
    __has_builtin(__builtin_amdgcn_s_wait_asynccnt)
#define USE_ASYNC_LDS 1
#endif
#endif

#ifdef USE_ASYNC_LDS
typedef __attribute__((ext_vector_type(4))) int int4v;
typedef __attribute__((address_space(1))) int4v g_int4;   // global int4
typedef __attribute__((address_space(3))) int4v l_int4;   // LDS int4
#endif

// ---------------------------------------------------------------- W2 -> f16
__global__ void w2_to_half(const float* __restrict__ W2, half_t* __restrict__ W2h, int n) {
    int idx = blockIdx.x * blockDim.x + threadIdx.x;
    if (idx < n) W2h[idx] = (half_t)W2[idx];
}

// ------------------------------------------- LayerNorm + proj (a/bb in f16)
// abuf[m][k]: m = pos*32 + c, k = r   (k contiguous)
// bbuf[n][k]: n = pos*32 + d, k = r
__global__ __launch_bounds__(64) void ln_proj(
    const float* __restrict__ m,    const float* __restrict__ ln_w,
    const float* __restrict__ ln_b, const float* __restrict__ W1,
    const float* __restrict__ b1,
    half_t* __restrict__ abuf, half_t* __restrict__ bbuf) {

    __shared__ float W1s[2 * DIM * IN_DIM];   // 32 KB
    __shared__ float rows[64 * 129];          // 33 KB, stride 129 = conflict-free

    const int tid = threadIdx.x;
    for (int t = tid; t < 2 * DIM * IN_DIM; t += 64) W1s[t] = W1[t];

    const int p   = blockIdx.x * 64 + tid;    // 0 .. 32767
    const int r   = p >> 8;
    const int pos = p & 255;
    const float* mrow = m + (size_t)(r * SEQ + pos) * IN_DIM;
    float* lrow = rows + tid * 129;

    float s = 0.f, ss = 0.f;
    for (int t = 0; t < IN_DIM; t += 4) {
        float4 v = *(const float4*)(mrow + t);
        lrow[t + 0] = v.x; lrow[t + 1] = v.y; lrow[t + 2] = v.z; lrow[t + 3] = v.w;
        s  += v.x + v.y + v.z + v.w;
        ss += v.x * v.x + v.y * v.y + v.z * v.z + v.w * v.w;
    }
    const float mu   = s * (1.0f / IN_DIM);
    const float var  = ss * (1.0f / IN_DIM) - mu * mu;
    const float rstd = rsqrtf(var + 1e-5f);
    for (int t = 0; t < IN_DIM; ++t)
        lrow[t] = (lrow[t] - mu) * rstd * ln_w[t] + ln_b[t];

    __syncthreads();   // W1s ready

    for (int f = 0; f < 2 * DIM; ++f) {
        float acc = b1[f];
        const float* wr = W1s + f * IN_DIM;
        #pragma unroll 8
        for (int t = 0; t < IN_DIM; ++t) acc += lrow[t] * wr[t];
        half_t h = (half_t)acc;
        if (f < DIM) abuf[(size_t)(pos * DIM + f)         * RROWS + r] = h;
        else         bbuf[(size_t)(pos * DIM + (f - DIM)) * RROWS + r] = h;
    }
}

// ----------------------- fused: (a x bb outer-product-mean) @ W2^T, all WMMA
__global__ __launch_bounds__(256) void fused_opm(
    const half_t* __restrict__ abuf, const half_t* __restrict__ bbuf,
    const half_t* __restrict__ W2h,  const float* __restrict__ b2,
    float* __restrict__ zout) {

    __shared__ __align__(16) half_t sh[2 * 128 * ASTRIDE];   // 68 KB
    half_t* Ash = sh;                  // [m][k] stride ASTRIDE
    half_t* Bsh = sh + 128 * ASTRIDE;  // [n][k] stride ASTRIDE
    half_t* Osh = sh;                  // reused after GEMM1: [pair][k2] stride OSTRIDE

    const int tid  = threadIdx.x;
    const int lane = tid & 31;
    const int wave = tid >> 5;
    const int i0 = blockIdx.y * TI;
    const int j0 = blockIdx.x * TJ;
    const int fcol = wave * 16 + (lane & 15);   // this wave's f-column in GEMM2

    // ---- stage GEMM1 operand tiles (each 128 rows x 128 k of f16 = 32 KB)
    const half_t* ga = abuf + (size_t)i0 * DIM * RROWS;
    const half_t* gb = bbuf + (size_t)j0 * DIM * RROWS;
    #pragma unroll
    for (int t = 0; t < 8; ++t) {
        int q   = tid + t * 256;       // 2048 16B chunks per tile
        int row = q >> 4;
        int off = (q & 15) * 8;
#ifdef USE_ASYNC_LDS
        __builtin_amdgcn_global_load_async_to_lds_b128(
            (g_int4*)(ga + row * IN_DIM + off),
            (l_int4*)(Ash + row * ASTRIDE + off), 0, 0);
        __builtin_amdgcn_global_load_async_to_lds_b128(
            (g_int4*)(gb + row * IN_DIM + off),
            (l_int4*)(Bsh + row * ASTRIDE + off), 0, 0);
#else
        *(v8h*)(Ash + row * ASTRIDE + off) = *(const v8h*)(ga + row * IN_DIM + off);
        *(v8h*)(Bsh + row * ASTRIDE + off) = *(const v8h*)(gb + row * IN_DIM + off);
#endif
    }
    // prefetch this wave's W2 f-tile rows (L2-resident) while GEMM1 runs
    #pragma unroll
    for (int off = 0; off < DIM * DIM; off += 128)
        __builtin_prefetch(W2h + (size_t)fcol * (DIM * DIM) + off, 0, 3);
#ifdef USE_ASYNC_LDS
    __builtin_amdgcn_s_wait_asynccnt(0);
#endif
    __syncthreads();

    // ---- GEMM1: 128x128 tile, K=128. wave w owns m-tile w, all 8 n-tiles.
    v8f acc[8] = {};
    const int arow  = wave * 16 + (lane & 15);        // A fragment: lane = M row
    const int ahalf = (lane >> 4) ? 8 : 0;            // K chunks {0-7,16-23} vs {8-15,24-31}
    const int bcol  = lane & 15;                      // B fragment: lane = N col
    const int bhalf = (lane >> 4) ? 16 : 0;           // K 0-15 vs 16-31

    for (int kk = 0; kk < 4; ++kk) {
        const int k0 = kk * 32;
        union { v16h v; v8h h[2]; } af;
        const half_t* ap = Ash + arow * ASTRIDE + k0 + ahalf;
        af.h[0] = *(const v8h*)(ap);
        af.h[1] = *(const v8h*)(ap + 16);
        #pragma unroll
        for (int nt = 0; nt < 8; ++nt) {
            union { v16h v; v8h h[2]; } bf;
            const half_t* bp = Bsh + (nt * 16 + bcol) * ASTRIDE + k0 + ahalf;
            bf.h[0] = *(const v8h*)(bp);
            bf.h[1] = *(const v8h*)(bp + 16);
            acc[nt] = __builtin_amdgcn_wmma_f32_16x16x32_f16(
                false, af.v, false, bf.v, (short)0, acc[nt], false, false);
        }
    }
    __syncthreads();   // all reads of Ash/Bsh complete before O overwrites

    // ---- repack O tile into LDS as GEMM2 A operand: 16 pairs x 1024, f16, /128
    const float scale = 1.0f / (float)RROWS;
    #pragma unroll
    for (int nt = 0; nt < 8; ++nt) {
        const int n  = nt * 16 + (lane & 15);
        const int jj = n >> 5, d = n & 31;
        #pragma unroll
        for (int e = 0; e < 8; ++e) {
            const int mrow = wave * 16 + ((lane < 16) ? e : (8 + e));
            const int ii = mrow >> 5, c = mrow & 31;
            Osh[(ii * TJ + jj) * OSTRIDE + (c * DIM + d)] = (half_t)(acc[nt][e] * scale);
        }
    }
    __syncthreads();

    // ---- GEMM2: M=16 pairs, N=128 (wave w -> f-tile w), K=1024
    v8f acc2 = {};
    const int prow = lane & 15;
    for (int k0 = 0; k0 < DIM * DIM; k0 += 32) {
        union { v16h v; v8h h[2]; } af;
        const half_t* ap = Osh + prow * OSTRIDE + k0 + ahalf;
        af.h[0] = *(const v8h*)(ap);
        af.h[1] = *(const v8h*)(ap + 16);
        union { v16h v; v8h h[2]; } bf;   // B col f = W2 row f (k contiguous in memory)
        const half_t* bp = W2h + (size_t)fcol * (DIM * DIM) + k0 + ahalf;
        bf.h[0] = *(const v8h*)(bp);
        bf.h[1] = *(const v8h*)(bp + 16);
        acc2 = __builtin_amdgcn_wmma_f32_16x16x32_f16(
            false, af.v, false, bf.v, (short)0, acc2, false, false);
    }

    // ---- epilogue: add b2, store z[i,j,f]
    const float bias = b2[fcol];
    #pragma unroll
    for (int e = 0; e < 8; ++e) {
        const int pair = (lane < 16) ? e : (8 + e);
        const int i = i0 + (pair >> 2);
        const int j = j0 + (pair & 3);
        zout[((size_t)i * SEQ + j) * IN_DIM + fcol] = acc2[e] + bias;
    }
}

// --------------------------------------------------------------------------
extern "C" void kernel_launch(void* const* d_in, const int* in_sizes, int n_in,
                              void* d_out, int out_size, void* d_ws, size_t ws_size,
                              hipStream_t stream) {
    (void)in_sizes; (void)n_in; (void)out_size; (void)ws_size;
    const float* m    = (const float*)d_in[0];
    const float* ln_w = (const float*)d_in[1];
    const float* ln_b = (const float*)d_in[2];
    const float* W1   = (const float*)d_in[3];
    const float* b1   = (const float*)d_in[4];
    const float* W2   = (const float*)d_in[5];
    const float* b2   = (const float*)d_in[6];
    float* z = (float*)d_out;

    char* ws = (char*)d_ws;
    half_t* abuf = (half_t*)ws;                                   // 8192x128 f16 = 2 MB
    half_t* bbuf = (half_t*)(ws + (size_t)8192 * 128 * 2);        // 2 MB
    half_t* W2h  = (half_t*)(ws + (size_t)2 * 8192 * 128 * 2);    // 256 KB

    const int nW2 = IN_DIM * DIM * DIM;
    w2_to_half<<<(nW2 + 255) / 256, 256, 0, stream>>>(W2, W2h, nW2);
    ln_proj<<<512, 64, 0, stream>>>(m, ln_w, ln_b, W1, b1, abuf, bbuf);
    dim3 grid(SEQ / TJ, SEQ / TI);
    fused_opm<<<grid, 256, 0, stream>>>(abuf, bbuf, W2h, b2, z);
}
